// DAWN_88124138979393
// MI455X (gfx1250) — matrix operand order repaired and checked by reading, not compiled, not run on hardware
//
#include <hip/hip_runtime.h>

// ---------------------------------------------------------------------------
// Problem constants (from reference)
// ---------------------------------------------------------------------------
#define BB   2
#define SS   2048
#define DD   1024
#define HH   16
#define RR_  64
#define NN   32
#define RKK  128
#define DHC  64
#define MM   (BB * SS)        // 4096 tokens
#define EPS_ 1e-5f

typedef __attribute__((ext_vector_type(16))) __bf16 v16bf;
typedef __attribute__((ext_vector_type(8)))  float  v8f;

// ---------------------------------------------------------------------------
// bf16 helpers (storage = unsigned short)
// ---------------------------------------------------------------------------
__device__ __forceinline__ __bf16 DAWN_as_bf16(unsigned short u) {
    union { unsigned short u; __bf16 b; } x; x.u = u; return x.b;
}
__device__ __forceinline__ unsigned short DAWN_f2bf(float f) {
    unsigned int u = __float_as_uint(f);
    u += 0x7FFFu + ((u >> 16) & 1u);          // round-to-nearest-even
    return (unsigned short)(u >> 16);
}

__device__ __forceinline__ v8f DAWN_wmma(v16bf a, v16bf b, v8f c) {
    // D = A(16x32 bf16) * B(32x16 bf16) + C(16x16 f32)
    return __builtin_amdgcn_wmma_f32_16x16x32_bf16(
        /*neg_a=*/false, a, /*neg_b=*/false, b,
        /*c_mod=*/(short)0, c, /*reuse_a=*/false, /*reuse_b=*/false);
}

// ---------------------------------------------------------------------------
// gfx1250 async global->LDS copy (ASYNCcnt path) + wait
// ---------------------------------------------------------------------------
__device__ __forceinline__ void DAWN_async_ld_b128(void* lds_ptr, const void* gptr) {
    unsigned lds_off = (unsigned)(unsigned long long)lds_ptr;
    unsigned long long ga = (unsigned long long)gptr;
    asm volatile("global_load_async_to_lds_b128 %0, %1, off"
                 :: "v"(lds_off), "v"(ga)
                 : "memory");
}
__device__ __forceinline__ void DAWN_wait_async() {
    asm volatile("s_wait_asynccnt 0x0" ::: "memory");
}

// ---------------------------------------------------------------------------
// Fragment loaders.
// ---------------------------------------------------------------------------
// A fragment: 16x32 tile at (row0, k0); lane l -> row l&15; K = (l>>4)*16 + i
// 16-byte aligned contiguous runs -> two b128 LDS loads
template <int STRIDE>
__device__ __forceinline__ v16bf DAWN_frag_a(const unsigned short* s, int row0, int k0, int lane) {
    const int hi = lane >> 4;
    const unsigned short* p = s + (size_t)(row0 + (lane & 15)) * STRIDE + k0 + hi * 16;
    union { v16bf v; uint4 q[2]; } x;
    x.q[0] = *(const uint4*)p;          // K = hi*16 + 0..7
    x.q[1] = *(const uint4*)(p + 8);    // K = hi*16 + 8..15
    return x.v;
}

// B fragment: 32x16 tile at (k0, col0); lane l -> col l&15; K = (l>>4)*16 + i
// (column-strided: u16 gathers; shared by two WMMAs per wave)
template <int STRIDE>
__device__ __forceinline__ v16bf DAWN_frag_b(const unsigned short* s, int k0, int col0, int lane) {
    const unsigned short* p = s + (size_t)(k0 + (lane >> 4) * 16) * STRIDE + col0 + (lane & 15);
    v16bf b;
#pragma unroll
    for (int i = 0; i < 16; ++i) b[i] = DAWN_as_bf16(p[(size_t)i * STRIDE]);
    return b;
}

// B fragment where logical B[k][n] = LDS[n][k]  (Q*K^T): contiguous in k
template <int STRIDE>
__device__ __forceinline__ v16bf DAWN_frag_bT(const unsigned short* s, int k0, int col0, int lane) {
    const unsigned short* p = s + (size_t)(col0 + (lane & 15)) * STRIDE + k0 + (lane >> 4) * 16;
    union { v16bf v; uint4 q[2]; } x;
    x.q[0] = *(const uint4*)p;
    x.q[1] = *(const uint4*)(p + 8);
    return x.v;
}

// ---------------------------------------------------------------------------
// LayerNorm (fp32 in) -> bf16 out
// ---------------------------------------------------------------------------
__global__ __launch_bounds__(256) void DAWN_ln_kernel(
    const float* __restrict__ x, const float* __restrict__ g,
    const float* __restrict__ bta, unsigned short* __restrict__ out)
{
    __shared__ float red[256];
    const int row = blockIdx.x;
    const int tid = threadIdx.x;
    const float* xr = x + (size_t)row * DD;
    float v[4]; float s = 0.f;
#pragma unroll
    for (int i = 0; i < 4; ++i) { v[i] = xr[tid + i * 256]; s += v[i]; }
    red[tid] = s; __syncthreads();
    for (int off = 128; off > 0; off >>= 1) { if (tid < off) red[tid] += red[tid + off]; __syncthreads(); }
    const float mu = red[0] * (1.0f / DD);
    __syncthreads();
    float vs = 0.f;
#pragma unroll
    for (int i = 0; i < 4; ++i) { float d = v[i] - mu; vs += d * d; }
    red[tid] = vs; __syncthreads();
    for (int off = 128; off > 0; off >>= 1) { if (tid < off) red[tid] += red[tid + off]; __syncthreads(); }
    const float rstd = rsqrtf(red[0] * (1.0f / DD) + EPS_);
#pragma unroll
    for (int i = 0; i < 4; ++i) {
        const int c = tid + i * 256;
        out[(size_t)row * DD + c] = DAWN_f2bf(g[c] * (v[i] - mu) * rstd + bta[c]);
    }
}

// ---------------------------------------------------------------------------
// Weight packing kernels
// ---------------------------------------------------------------------------
// f_* [N,D,Rr] -> bf16 [D, N*Rr]
__global__ void DAWN_pack_feature(const float* __restrict__ src,
                                  unsigned short* __restrict__ dst, int Rr)
{
    const int total = DD * NN * Rr;
    const int idx = blockIdx.x * 256 + threadIdx.x;
    if (idx >= total) return;
    const int r = idx % Rr;
    const int t = idx / Rr;
    const int n = t % NN;
    const int d = t / NN;
    dst[idx] = DAWN_f2bf(src[((size_t)n * DD + d) * Rr + r]);  // dst[d][n*Rr+r] == dst[idx]
}

// contiguous fp32 -> bf16
__global__ void DAWN_cvt_bf16(const float* __restrict__ src,
                              unsigned short* __restrict__ dst, int nElem)
{
    const int i = blockIdx.x * 256 + threadIdx.x;
    if (i < nElem) dst[i] = DAWN_f2bf(src[i]);
}

// W_O [D,D] -> bf16 transpose: dst[k*D + c] = W_O[c*D + k]
__global__ void DAWN_pack_wo(const float* __restrict__ src, unsigned short* __restrict__ dst)
{
    const int i = blockIdx.x * 256 + threadIdx.x;  // i = k*D + c
    const int c = i % DD, k = i / DD;
    dst[i] = DAWN_f2bf(src[(size_t)c * DD + k]);
}

// ---------------------------------------------------------------------------
// h[m,r] = sum_n w[m,n] * Hall[m, n*Rr + r]
// ---------------------------------------------------------------------------
__global__ void DAWN_reduce_h(const float* __restrict__ Hall, const float* __restrict__ w,
                              float* __restrict__ h, int Rr)
{
    const int m = blockIdx.x;
    const int r = threadIdx.x;                 // blockDim.x == Rr
    const float* Hm = Hall + (size_t)m * (NN * Rr) + r;
    const float* wm = w + (size_t)m * NN;
    float acc = 0.f;
#pragma unroll 8
    for (int n = 0; n < NN; ++n) acc += wm[n] * Hm[(size_t)n * Rr];
    h[(size_t)m * Rr + r] = acc;
}

// T[m, n*Rr + r] = bf16( w[m,n] * h[m,r] )
__global__ void DAWN_build_t(const float* __restrict__ h, const float* __restrict__ w,
                             unsigned short* __restrict__ T, int Rr)
{
    const int NR = NN * Rr;
    const int idx = blockIdx.x * 256 + threadIdx.x;
    const int m = idx / NR;
    const int c = idx % NR;
    const int n = c / Rr;
    const int r = c % Rr;
    T[idx] = DAWN_f2bf(w[(size_t)m * NN + n] * h[(size_t)m * Rr + r]);
}

// ---------------------------------------------------------------------------
// bf16 WMMA GEMM: C[M,Nc] = A[M,K] * B[K,Nc]   (A,B bf16 row-major, C fp32)
//   64x64 block tile, K step 32, 8 wave32 waves.
//   Each wave: 32 rows x 16 cols (two A frags share one B frag -> half the
//   column-strided LDS gathers).  Tiles staged via async LDS copies.
// Epilogue modes:
//   0: Cf[m*Nc+c] = acc
//   1: bf16 heads layout  Cbf[((b*H+head)*S + s)*DH + wi] = bf16(acc*scale)
//   2: Cf[m*Nc+c] = resid[m*Nc+c] + acc
// ---------------------------------------------------------------------------
#define A_STR 40
#define B_STR 72
__global__ __launch_bounds__(256) void DAWN_gemm_bf16(
    const unsigned short* __restrict__ A, const unsigned short* __restrict__ Bm,
    int Mdim, int Ndim, int Kdim,
    float* __restrict__ Cf, unsigned short* __restrict__ Cbf,
    const float* __restrict__ resid, int mode, float scale)
{
    __shared__ __attribute__((aligned(16))) unsigned short As[64 * A_STR];
    __shared__ __attribute__((aligned(16))) unsigned short Bs[32 * B_STR];
    const int tid = threadIdx.x, lane = tid & 31, wave = tid >> 5;
    const int m0 = blockIdx.y * 64, n0 = blockIdx.x * 64;
    const int rh = wave >> 2;  // 32-row half
    const int ct = wave & 3;   // 16-col tile
    v8f acc0 = {}, acc1 = {};

    const int ra = tid >> 2, ca = (tid & 3) * 8;   // A: 64x32, 8 u16/thread
    const int rb = tid >> 3, cb = (tid & 7) * 8;   // B: 32x64, 8 u16/thread
    const unsigned short* ga = A  + (size_t)(m0 + ra) * Kdim + ca;
    const unsigned short* gb = Bm + (size_t)rb * Ndim + n0 + cb;
    const size_t gbStep = (size_t)32 * Ndim;

    for (int k0 = 0; k0 < Kdim; k0 += 32) {
        DAWN_async_ld_b128(&As[ra * A_STR + ca], ga);
        DAWN_async_ld_b128(&Bs[rb * B_STR + cb], gb);
        __builtin_prefetch(ga + 32, 0, 1);          // speculative; dropped if OOB
        __builtin_prefetch(gb + gbStep, 0, 1);
        DAWN_wait_async();
        __syncthreads();
        v16bf a0 = DAWN_frag_a<A_STR>(As, rh * 32,      0, lane);
        v16bf a1 = DAWN_frag_a<A_STR>(As, rh * 32 + 16, 0, lane);
        v16bf b  = DAWN_frag_b<B_STR>(Bs, 0, ct * 16, lane);
        acc0 = DAWN_wmma(a0, b, acc0);
        acc1 = DAWN_wmma(a1, b, acc1);
        __syncthreads();
        ga += 32;
        gb += gbStep;
    }

    const int r0 = m0 + rh * 32 + (lane >> 4) * 8;     // acc0 rows r0+j, acc1 rows r0+16+j
    const int c0 = n0 + ct * 16 + (lane & 15);
    if (mode == 0) {
#pragma unroll
        for (int j = 0; j < 8; ++j) {
            Cf[(size_t)(r0 + j)      * Ndim + c0] = acc0[j];
            Cf[(size_t)(r0 + 16 + j) * Ndim + c0] = acc1[j];
        }
    } else if (mode == 1) {
        const int head = c0 >> 6, wi = c0 & 63;
#pragma unroll
        for (int j = 0; j < 8; ++j) {
            int row = r0 + j;
            Cbf[(((size_t)((row >> 11) * HH + head) * SS) + (row & (SS - 1))) * DHC + wi]
                = DAWN_f2bf(acc0[j] * scale);
            row = r0 + 16 + j;
            Cbf[(((size_t)((row >> 11) * HH + head) * SS) + (row & (SS - 1))) * DHC + wi]
                = DAWN_f2bf(acc1[j] * scale);
        }
    } else {
#pragma unroll
        for (int j = 0; j < 8; ++j) {
            const size_t ro0 = (size_t)(r0 + j) * Ndim + c0;
            const size_t ro1 = (size_t)(r0 + 16 + j) * Ndim + c0;
            Cf[ro0] = resid[ro0] + acc0[j];
            Cf[ro1] = resid[ro1] + acc1[j];
        }
    }
}

// ---------------------------------------------------------------------------
// Flash attention (causal), bf16 QKV [BH, S, DH], out bf16 [B,S,D] row layout.
// One block = 64 queries of one (b,h); 8 waves, each 32 rows x 16 cols.
// ---------------------------------------------------------------------------
#define Q_STR 72
__global__ __launch_bounds__(256) void DAWN_attn(
    const unsigned short* __restrict__ Qh, const unsigned short* __restrict__ Kh,
    const unsigned short* __restrict__ Vh, unsigned short* __restrict__ Out)
{
    __shared__ __attribute__((aligned(16))) unsigned short Qs[64 * Q_STR];
    __shared__ __attribute__((aligned(16))) unsigned short Ks[64 * Q_STR];
    __shared__ __attribute__((aligned(16))) unsigned short Vs[64 * Q_STR];
    __shared__ __attribute__((aligned(16))) unsigned short Ps[64 * Q_STR];
    __shared__ float Sbuf[64 * 68];
    __shared__ float mrow[64], lrow[64], arow[64];

    const int tid = threadIdx.x, lane = tid & 31, wave = tid >> 5;
    const int q0 = blockIdx.x * 64;
    const int bh = blockIdx.y;
    const size_t base = (size_t)bh * SS * DHC;
    const int rh = wave >> 2, ct = wave & 3;

    const int rt = tid >> 2, ctb = (tid & 3) * 16;  // 64x64 tile staging pattern
    {   // load Q tile 64x64 (async)
        const unsigned short* g = Qh + base + (size_t)(q0 + rt) * DHC + ctb;
        DAWN_async_ld_b128(&Qs[rt * Q_STR + ctb],     g);
        DAWN_async_ld_b128(&Qs[rt * Q_STR + ctb + 8], g + 8);
    }
    if (tid < 64) { mrow[tid] = -1e30f; lrow[tid] = 0.f; }
    v8f o0 = {}, o1 = {};
    DAWN_wait_async();
    __syncthreads();

    const unsigned short* gk = Kh + base + (size_t)rt * DHC + ctb;
    const unsigned short* gv = Vh + base + (size_t)rt * DHC + ctb;
    const size_t kvStep = (size_t)64 * DHC;

    const int rl = rh * 32 + (lane >> 4) * 8;        // frag row base (j adds 0..7)
    const int cl = ct * 16 + (lane & 15);            // frag col

    for (int j0 = 0; j0 <= q0; j0 += 64) {
        DAWN_async_ld_b128(&Ks[rt * Q_STR + ctb],     gk);
        DAWN_async_ld_b128(&Ks[rt * Q_STR + ctb + 8], gk + 8);
        DAWN_async_ld_b128(&Vs[rt * Q_STR + ctb],     gv);
        DAWN_async_ld_b128(&Vs[rt * Q_STR + ctb + 8], gv + 8);
        DAWN_wait_async();
        __syncthreads();

        // S = Q * K^T (32 rows x 16 keys per wave), scale folded into Q
        v8f s0 = {}, s1 = {};
#pragma unroll
        for (int kk = 0; kk < DHC; kk += 32) {
            v16bf aq0 = DAWN_frag_a<Q_STR>(Qs, rh * 32,      kk, lane);
            v16bf aq1 = DAWN_frag_a<Q_STR>(Qs, rh * 32 + 16, kk, lane);
            v16bf bk  = DAWN_frag_bT<Q_STR>(Ks, kk, ct * 16, lane);
            s0 = DAWN_wmma(aq0, bk, s0);
            s1 = DAWN_wmma(aq1, bk, s1);
        }
#pragma unroll
        for (int j = 0; j < 8; ++j) {           // masked store of scores
            const int rr0 = rl + j, rr1 = rl + 16 + j;
            Sbuf[rr0 * 68 + cl] = (j0 + cl <= q0 + rr0) ? s0[j] : -1e30f;
            Sbuf[rr1 * 68 + cl] = (j0 + cl <= q0 + rr1) ? s1[j] : -1e30f;
        }
        __syncthreads();

        if (tid < 64) {                          // online softmax, one row/thread
            const float mold = mrow[tid];
            float mx = mold;
            for (int c = 0; c < 64; ++c) mx = fmaxf(mx, Sbuf[tid * 68 + c]);
            const float al = __expf(mold - mx);
            float sum = 0.f;
            for (int c = 0; c < 64; ++c) {
                const float p = __expf(Sbuf[tid * 68 + c] - mx);
                sum += p;
                Ps[tid * Q_STR + c] = DAWN_f2bf(p);
            }
            mrow[tid] = mx;
            lrow[tid] = lrow[tid] * al + sum;
            arow[tid] = al;
        }
        __syncthreads();

#pragma unroll
        for (int j = 0; j < 8; ++j) {            // rescale running O
            o0[j] *= arow[rl + j];
            o1[j] *= arow[rl + 16 + j];
        }
#pragma unroll
        for (int kk = 0; kk < 64; kk += 32) {    // O += P * V
            v16bf ap0 = DAWN_frag_a<Q_STR>(Ps, rh * 32,      kk, lane);
            v16bf ap1 = DAWN_frag_a<Q_STR>(Ps, rh * 32 + 16, kk, lane);
            v16bf bv  = DAWN_frag_b<Q_STR>(Vs, kk, ct * 16, lane);
            o0 = DAWN_wmma(ap0, bv, o0);
            o1 = DAWN_wmma(ap1, bv, o1);
        }
        __syncthreads();
        gk += kvStep;
        gv += kvStep;
    }

    const int b = bh >> 4, hh = bh & 15;
#pragma unroll
    for (int j = 0; j < 8; ++j) {
        const int rr0 = rl + j, rr1 = rl + 16 + j;
        Out[((size_t)(b * SS + q0 + rr0)) * DD + hh * DHC + cl]
            = DAWN_f2bf(o0[j] * (1.0f / lrow[rr0]));
        Out[((size_t)(b * SS + q0 + rr1)) * DD + hh * DHC + cl]
            = DAWN_f2bf(o1[j] * (1.0f / lrow[rr1]));
    }
}

// ---------------------------------------------------------------------------
// Host orchestration
// ---------------------------------------------------------------------------
extern "C" void kernel_launch(void* const* d_in, const int* in_sizes, int n_in,
                              void* d_out, int out_size, void* d_ws, size_t ws_size,
                              hipStream_t stream)
{
    (void)in_sizes; (void)n_in; (void)out_size; (void)ws_size;
    const float* x        = (const float*)d_in[0];
    const float* f_qk     = (const float*)d_in[1];
    const float* f_v      = (const float*)d_in[2];
    const float* r_qk     = (const float*)d_in[3];
    const float* r_v      = (const float*)d_in[4];
    const float* f_know   = (const float*)d_in[5];
    const float* r_know   = (const float*)d_in[6];
    const float* W_O      = (const float*)d_in[7];
    const float* gamma1   = (const float*)d_in[8];
    const float* beta1    = (const float*)d_in[9];
    const float* gamma2   = (const float*)d_in[10];
    const float* beta2    = (const float*)d_in[11];
    const float* w_fq     = (const float*)d_in[12];
    const float* w_fk     = (const float*)d_in[13];
    const float* w_fv     = (const float*)d_in[14];
    const float* w_rq     = (const float*)d_in[15];
    const float* w_rk     = (const float*)d_in[16];
    const float* w_rv     = (const float*)d_in[17];
    const float* w_know_f = (const float*)d_in[18];
    const float* w_know_r = (const float*)d_in[19];
    float* out = (float*)d_out;

    char* wsb = (char*)d_ws;
    size_t off = 0;
    auto take = [&](size_t bytes) -> char* {
        char* p = wsb + off;
        off += (bytes + 255) & ~(size_t)255;
        return p;
    };
    unsigned short* nx    = (unsigned short*)take((size_t)MM * DD * 2);           // LN out (reused for LN2)
    unsigned short* wpkA  = (unsigned short*)take((size_t)DD * NN * RR_ * 2);     // f_qk pack
    unsigned short* wpkB  = (unsigned short*)take((size_t)DD * NN * RR_ * 2);     // f_v pack
    float*          Hbuf  = (float*)take((size_t)MM * NN * RKK * 4);              // widest H (64 MB)
    float*          h_q   = (float*)take((size_t)MM * RR_ * 4);
    float*          h_k   = (float*)take((size_t)MM * RR_ * 4);
    float*          h_v   = (float*)take((size_t)MM * RR_ * 4);
    float*          h_kn  = (float*)take((size_t)MM * RKK * 4);
    unsigned short* rpkQK = (unsigned short*)take((size_t)NN * RR_ * DD * 2);
    unsigned short* rpkV  = (unsigned short*)take((size_t)NN * RR_ * DD * 2);
    unsigned short* Tbuf  = (unsigned short*)take((size_t)MM * NN * RKK * 2);     // widest T (32 MB)
    unsigned short* Qh    = (unsigned short*)take((size_t)BB * HH * SS * DHC * 2);
    unsigned short* Kh    = (unsigned short*)take((size_t)BB * HH * SS * DHC * 2);
    unsigned short* Vh    = (unsigned short*)take((size_t)BB * HH * SS * DHC * 2);
    unsigned short* Abuf  = (unsigned short*)take((size_t)MM * DD * 2);
    unsigned short* wopk  = (unsigned short*)take((size_t)DD * DD * 2);
    float*          x1    = (float*)take((size_t)MM * DD * 4);
    unsigned short* wpkK  = (unsigned short*)take((size_t)DD * NN * RKK * 2);
    unsigned short* rpkK  = (unsigned short*)take((size_t)NN * RKK * DD * 2);

    const dim3 blk(256);
    auto gemm = [&](const unsigned short* A, const unsigned short* Bm,
                    int Md, int Nd, int Kd, float* Cf, unsigned short* Cbf,
                    const float* resid, int mode, float scale) {
        dim3 grid(Nd / 64, Md / 64);
        DAWN_gemm_bf16<<<grid, blk, 0, stream>>>(A, Bm, Md, Nd, Kd, Cf, Cbf, resid, mode, scale);
    };

    // ---- attention circuit ----
    DAWN_ln_kernel<<<MM, blk, 0, stream>>>(x, gamma1, beta1, nx);
    DAWN_pack_feature<<<(DD * NN * RR_) / 256, blk, 0, stream>>>(f_qk, wpkA, RR_);
    DAWN_pack_feature<<<(DD * NN * RR_) / 256, blk, 0, stream>>>(f_v, wpkB, RR_);

    gemm(nx, wpkA, MM, NN * RR_, DD, Hbuf, nullptr, nullptr, 0, 1.0f);            // H_qk
    DAWN_reduce_h<<<MM, RR_, 0, stream>>>(Hbuf, w_fq, h_q, RR_);
    DAWN_reduce_h<<<MM, RR_, 0, stream>>>(Hbuf, w_fk, h_k, RR_);
    gemm(nx, wpkB, MM, NN * RR_, DD, Hbuf, nullptr, nullptr, 0, 1.0f);            // H_v
    DAWN_reduce_h<<<MM, RR_, 0, stream>>>(Hbuf, w_fv, h_v, RR_);

    DAWN_cvt_bf16<<<(NN * RR_ * DD) / 256, blk, 0, stream>>>(r_qk, rpkQK, NN * RR_ * DD);
    DAWN_cvt_bf16<<<(NN * RR_ * DD) / 256, blk, 0, stream>>>(r_v, rpkV, NN * RR_ * DD);

    const float qscale = 0.125f;                                                  // 1/sqrt(DH)
    DAWN_build_t<<<((size_t)MM * NN * RR_) / 256, blk, 0, stream>>>(h_q, w_rq, Tbuf, RR_);
    gemm(Tbuf, rpkQK, MM, DD, NN * RR_, nullptr, Qh, nullptr, 1, qscale);
    DAWN_build_t<<<((size_t)MM * NN * RR_) / 256, blk, 0, stream>>>(h_k, w_rk, Tbuf, RR_);
    gemm(Tbuf, rpkQK, MM, DD, NN * RR_, nullptr, Kh, nullptr, 1, 1.0f);
    DAWN_build_t<<<((size_t)MM * NN * RR_) / 256, blk, 0, stream>>>(h_v, w_rv, Tbuf, RR_);
    gemm(Tbuf, rpkV, MM, DD, NN * RR_, nullptr, Vh, nullptr, 1, 1.0f);

    DAWN_attn<<<dim3(SS / 64, BB * HH), blk, 0, stream>>>(Qh, Kh, Vh, Abuf);

    DAWN_pack_wo<<<(DD * DD) / 256, blk, 0, stream>>>(W_O, wopk);
    gemm(Abuf, wopk, MM, DD, DD, x1, nullptr, x, 2, 1.0f);                        // x1 = x + attn@W_O^T

    // ---- knowledge circuit ----
    DAWN_ln_kernel<<<MM, blk, 0, stream>>>(x1, gamma2, beta2, nx);
    DAWN_pack_feature<<<(DD * NN * RKK) / 256, blk, 0, stream>>>(f_know, wpkK, RKK);
    gemm(nx, wpkK, MM, NN * RKK, DD, Hbuf, nullptr, nullptr, 0, 1.0f);            // H_know
    DAWN_reduce_h<<<MM, RKK, 0, stream>>>(Hbuf, w_know_f, h_kn, RKK);
    DAWN_cvt_bf16<<<(NN * RKK * DD) / 256, blk, 0, stream>>>(r_know, rpkK, NN * RKK * DD);
    DAWN_build_t<<<((size_t)MM * NN * RKK) / 256, blk, 0, stream>>>(h_kn, w_know_r, Tbuf, RKK);
    gemm(Tbuf, rpkK, MM, DD, NN * RKK, out, nullptr, x1, 2, 1.0f);                // out = x1 + know_out
}
